// SelfAttention_309237646145
// MI455X (gfx1250) — matrix operand order repaired and checked by reading, not compiled, not run on hardware
//
#include <hip/hip_runtime.h>
#include <hip/hip_bf16.h>

// ---------- types ----------
typedef __bf16 bf16_t;
typedef __attribute__((ext_vector_type(16))) __bf16 v16bf;
typedef __attribute__((ext_vector_type(8)))  __bf16 v8bf;
typedef __attribute__((ext_vector_type(8)))  float  v8f;

#define N_HEADS 12
#define D_EMBED 768
#define BATCH   2
#define SEQ     2048
#define DH      64
#define D3      (3 * D_EMBED)
#define M_TOT   (BATCH * SEQ)      // 4096 rows of X
#define KCHUNK  32                 // keys staged per LDS chunk
#define QTILE   64                 // queries per attention workgroup (4 waves x 16)

__device__ __forceinline__ v16bf cat8(v8bf lo, v8bf hi) {
  return __builtin_shufflevector(lo, hi, 0,1,2,3,4,5,6,7,8,9,10,11,12,13,14,15);
}
__device__ __forceinline__ v8f wmma_bf16(v16bf a, v16bf b, v8f c) {
  // D = A(16x32) * B(32x16) + C, fp32 accumulate
  return __builtin_amdgcn_wmma_f32_16x16x32_bf16(false, a, false, b, (short)0, c, false, false);
}

// CDNA5 async copy: global -> LDS, 16B per lane, tracked by ASYNCcnt.
// lds_off = LDS byte address (low 32 bits of the generic pointer to a __shared__ object).
__device__ __forceinline__ void async_copy_b128(unsigned int lds_off, const bf16_t* g) {
  asm volatile("global_load_async_to_lds_b128 %0, %1, off"
               :: "v"(lds_off), "v"(g) : "memory");
}
__device__ __forceinline__ unsigned int lds_addr(const void* p) {
  return (unsigned int)(unsigned long long)p;
}

// ---------- conversion kernels ----------
__global__ void cvt_bf16(const float* __restrict__ src, bf16_t* __restrict__ dst, int n) {
  int i = blockIdx.x * blockDim.x + threadIdx.x;
  if (i < n) dst[i] = (bf16_t)src[i];
}

// src[K][N] fp32 -> dst[N][K] bf16 (column-major weight for contiguous B-fragments)
__global__ void cvt_bf16_tr(const float* __restrict__ src, bf16_t* __restrict__ dst, int K, int N) {
  int i = blockIdx.x * blockDim.x + threadIdx.x;
  if (i < K * N) {
    int k = i / N, n = i % N;
    dst[(size_t)n * K + k] = (bf16_t)src[i];
  }
}

// ---------- QKV projection: [4096,768] x [768,2304] + b -> Q,K (head-major), V (transposed) ----------
// One wave computes a 32x64 output tile: 2 (M) x 4 (N) WMMA fragments.
__global__ void __launch_bounds__(32)
qkv_gemm(const bf16_t* __restrict__ Xh, const bf16_t* __restrict__ WiT,
         const float* __restrict__ bin,
         bf16_t* __restrict__ Qh, bf16_t* __restrict__ Kh, bf16_t* __restrict__ Vt) {
  const int lane = threadIdx.x;
  const int hf = lane >> 4, lm = lane & 15;
  const int m0 = blockIdx.x * 32;
  const int n0 = blockIdx.y * 64;

  v8f acc[2][4] = {};
  const bf16_t* arow[2];
  arow[0] = Xh + (size_t)(m0 + lm) * D_EMBED;
  arow[1] = Xh + (size_t)(m0 + 16 + lm) * D_EMBED;
  const bf16_t* brow[4];
#pragma unroll
  for (int ni = 0; ni < 4; ++ni)
    brow[ni] = WiT + (size_t)(n0 + ni * 16 + lm) * D_EMBED + 16 * hf;

  for (int k0 = 0; k0 < D_EMBED; k0 += 32) {
    __builtin_prefetch(arow[0] + k0 + 128, 0, 3);
    v16bf a0 = cat8(*(const v8bf*)(arow[0] + k0 + 16 * hf), *(const v8bf*)(arow[0] + k0 + 16 * hf + 8));
    v16bf a1 = cat8(*(const v8bf*)(arow[1] + k0 + 16 * hf), *(const v8bf*)(arow[1] + k0 + 16 * hf + 8));
#pragma unroll
    for (int ni = 0; ni < 4; ++ni) {
      v16bf bfr = *(const v16bf*)(brow[ni] + k0);
      acc[0][ni] = wmma_bf16(a0, bfr, acc[0][ni]);
      acc[1][ni] = wmma_bf16(a1, bfr, acc[1][ni]);
    }
  }

#pragma unroll
  for (int mi = 0; mi < 2; ++mi) {
#pragma unroll
    for (int ni = 0; ni < 4; ++ni) {
#pragma unroll
      for (int r = 0; r < 8; ++r) {
        int m = m0 + mi * 16 + r + 8 * hf;     // global row (b*S + s)
        int n = n0 + ni * 16 + lm;             // column in [0, 2304)
        float val = acc[mi][ni][r] + bin[n];
        bf16_t bv = (bf16_t)val;
        int bb = m >> 11, s = m & (SEQ - 1);
        int which = n / D_EMBED, d = n % D_EMBED;
        int h = d >> 6, dd = d & 63;
        size_t hb = (size_t)(bb * N_HEADS + h);
        if (which == 0)      Qh[(hb * SEQ + s) * DH + dd] = bv;
        else if (which == 1) Kh[(hb * SEQ + s) * DH + dd] = bv;
        else                 Vt[(hb * DH + dd) * SEQ + s] = bv;  // V stored [bh][dh][s]
      }
    }
  }
}

// ---------- flash attention ----------
// 128-thread blocks: 4 waves x 16 queries = 64 queries. Each 32-key K/V chunk is staged
// in LDS by GLOBAL_LOAD_ASYNC_TO_LDS_B128 with double buffering: the copy for chunk c+1
// runs in the background while all 4 waves compute on chunk c. ASYNCcnt retires in order,
// so s_wait_asynccnt 4 == "oldest chunk (4 ops/thread) has landed".
__global__ void __launch_bounds__(128)
attn(const bf16_t* __restrict__ Qh, const bf16_t* __restrict__ Kh,
     const bf16_t* __restrict__ Vt, bf16_t* __restrict__ Ob) {
  __shared__ alignas(32) bf16_t Kbuf[2][KCHUNK * DH];   // [key][dh]  2 x 4KB
  __shared__ alignas(32) bf16_t Vbuf[2][DH * KCHUNK];   // [dh][key]  2 x 4KB
  __shared__ alignas(32) bf16_t Pl[4][16 * 32];         // per-wave P tile relayout

  const int tid  = threadIdx.x;
  const int wave = tid >> 5;
  const int lane = tid & 31;
  const int hf = lane >> 4, lm = lane & 15;
  const int q0 = blockIdx.x * QTILE + wave * 16;
  const int bh = blockIdx.y;
  const int b = bh / N_HEADS, h = bh % N_HEADS;

  const bf16_t* Qp = Qh + (size_t)bh * SEQ * DH;
  const bf16_t* Kp = Kh + (size_t)bh * SEQ * DH;
  const bf16_t* Vp = Vt + (size_t)bh * DH * SEQ;

  // per-thread cooperative copy coordinates
  const int vr = tid >> 1, vc = (tid & 1) * 16;

  // issue one chunk's async copies (4 x b128 per thread: 2 for K, 2 for V)
  auto issue_chunk = [&](int key0, int buf) {
    const bf16_t* ks = Kp + (size_t)key0 * DH + tid * 16;
    async_copy_b128(lds_addr(&Kbuf[buf][tid * 16]),          ks);
    async_copy_b128(lds_addr(&Kbuf[buf][tid * 16 + 8]),      ks + 8);
    const bf16_t* vs = Vp + (size_t)vr * SEQ + key0 + vc;
    async_copy_b128(lds_addr(&Vbuf[buf][vr * KCHUNK + vc]),     vs);
    async_copy_b128(lds_addr(&Vbuf[buf][vr * KCHUNK + vc + 8]), vs + 8);
  };

  // Q A-fragments (16 rows x 64), held in registers across the key loop
  const bf16_t* qrow = Qp + (size_t)(q0 + lm) * DH;
  v16bf aq0 = cat8(*(const v8bf*)(qrow + 16 * hf),      *(const v8bf*)(qrow + 16 * hf + 8));
  v16bf aq1 = cat8(*(const v8bf*)(qrow + 32 + 16 * hf), *(const v8bf*)(qrow + 40 + 16 * hf));

  v8f o0 = {}, o1 = {}, o2 = {}, o3 = {};            // 16 x 64 fp32 output accumulator
  float mrun[8], lrun[8];
#pragma unroll
  for (int r = 0; r < 8; ++r) { mrun[r] = -1e30f; lrun[r] = 0.0f; }
  const float scale = 0.125f;                        // 1/sqrt(64)

  const int NC = SEQ / KCHUNK;                       // 64 chunks
  issue_chunk(0, 0);
  issue_chunk(KCHUNK, 1);

  for (int c = 0; c < NC; ++c) {
    const int cur = c & 1;

    // retire the copy for chunk c (4 async ops/thread in flight per pending chunk)
    if (c < NC - 1) asm volatile("s_wait_asynccnt 4" ::: "memory");
    else            asm volatile("s_wait_asynccnt 0" ::: "memory");
    __syncthreads();

    // ---- scores S = Q * K^T (16 x 32 tile per wave), K fragments from LDS ----
    v8f s0v = {}, s1v = {};
    {
      const bf16_t* kb0 = &Kbuf[cur][lm * DH + 16 * hf];
      const bf16_t* kb1 = &Kbuf[cur][(16 + lm) * DH + 16 * hf];
      s0v = wmma_bf16(aq0, *(const v16bf*)(kb0),      s0v);
      s0v = wmma_bf16(aq1, *(const v16bf*)(kb0 + 32), s0v);
      s1v = wmma_bf16(aq0, *(const v16bf*)(kb1),      s1v);
      s1v = wmma_bf16(aq1, *(const v16bf*)(kb1 + 32), s1v);
    }

    // ---- online softmax per row (rows r+8*hf; 16 cols live in one 16-lane group) ----
    float corr[8];
    bf16_t* plw = Pl[wave];
#pragma unroll
    for (int r = 0; r < 8; ++r) {
      float s0 = s0v[r] * scale;
      float s1 = s1v[r] * scale;
      float mx = fmaxf(s0, s1);
      mx = fmaxf(mx, __shfl_xor(mx, 1));
      mx = fmaxf(mx, __shfl_xor(mx, 2));
      mx = fmaxf(mx, __shfl_xor(mx, 4));
      mx = fmaxf(mx, __shfl_xor(mx, 8));
      float mnew = fmaxf(mrun[r], mx);
      float cr = __expf(mrun[r] - mnew);
      corr[r] = cr;
      mrun[r] = mnew;
      float p0 = __expf(s0 - mnew);
      float p1 = __expf(s1 - mnew);
      float ps = p0 + p1;
      ps += __shfl_xor(ps, 1);
      ps += __shfl_xor(ps, 2);
      ps += __shfl_xor(ps, 4);
      ps += __shfl_xor(ps, 8);
      lrun[r] = lrun[r] * cr + ps;
      int row = r + 8 * hf;
      plw[row * 32 + lm]      = (bf16_t)p0;
      plw[row * 32 + 16 + lm] = (bf16_t)p1;
    }
#pragma unroll
    for (int r = 0; r < 8; ++r) {
      o0[r] *= corr[r]; o1[r] *= corr[r]; o2[r] *= corr[r]; o3[r] *= corr[r];
    }

    // LDS writes -> cross-lane reads within the wave: fence the DS counter
    asm volatile("s_wait_dscnt 0" ::: "memory");

    // ---- P as A-fragment (16 x 32), V chunk as four B-fragments from LDS ----
    const bf16_t* pp = plw + lm * 32;
    v16bf ap = cat8(*(const v8bf*)(pp + 16 * hf), *(const v8bf*)(pp + 16 * hf + 8));
    const bf16_t* vb = &Vbuf[cur][lm * KCHUNK + 16 * hf];
    o0 = wmma_bf16(ap, *(const v16bf*)(vb),               o0);
    o1 = wmma_bf16(ap, *(const v16bf*)(vb + 16 * KCHUNK), o1);
    o2 = wmma_bf16(ap, *(const v16bf*)(vb + 32 * KCHUNK), o2);
    o3 = wmma_bf16(ap, *(const v16bf*)(vb + 48 * KCHUNK), o3);

    __syncthreads();   // all waves done with buf[cur] before it is refilled
    if (c + 2 < NC) issue_chunk((c + 2) * KCHUNK, cur);
  }

  // ---- normalize and write merged-head output [B,S,D] in bf16 ----
#pragma unroll
  for (int r = 0; r < 8; ++r) {
    float inv = 1.0f / lrun[r];
    int s = q0 + r + 8 * hf;
    size_t base = ((size_t)b * SEQ + s) * D_EMBED + h * DH;
    Ob[base + lm]      = (bf16_t)(o0[r] * inv);
    Ob[base + 16 + lm] = (bf16_t)(o1[r] * inv);
    Ob[base + 32 + lm] = (bf16_t)(o2[r] * inv);
    Ob[base + 48 + lm] = (bf16_t)(o3[r] * inv);
  }
}

// ---------- output projection: [4096,768] x [768,768] + b -> fp32 ----------
__global__ void __launch_bounds__(32)
out_gemm(const bf16_t* __restrict__ Ob, const bf16_t* __restrict__ WoT,
         const float* __restrict__ bout, float* __restrict__ out) {
  const int lane = threadIdx.x;
  const int hf = lane >> 4, lm = lane & 15;
  const int m0 = blockIdx.x * 32;
  const int n0 = blockIdx.y * 64;

  v8f acc[2][4] = {};
  const bf16_t* arow[2];
  arow[0] = Ob + (size_t)(m0 + lm) * D_EMBED;
  arow[1] = Ob + (size_t)(m0 + 16 + lm) * D_EMBED;
  const bf16_t* brow[4];
#pragma unroll
  for (int ni = 0; ni < 4; ++ni)
    brow[ni] = WoT + (size_t)(n0 + ni * 16 + lm) * D_EMBED + 16 * hf;

  for (int k0 = 0; k0 < D_EMBED; k0 += 32) {
    __builtin_prefetch(arow[0] + k0 + 128, 0, 3);
    v16bf a0 = cat8(*(const v8bf*)(arow[0] + k0 + 16 * hf), *(const v8bf*)(arow[0] + k0 + 16 * hf + 8));
    v16bf a1 = cat8(*(const v8bf*)(arow[1] + k0 + 16 * hf), *(const v8bf*)(arow[1] + k0 + 16 * hf + 8));
#pragma unroll
    for (int ni = 0; ni < 4; ++ni) {
      v16bf bfr = *(const v16bf*)(brow[ni] + k0);
      acc[0][ni] = wmma_bf16(a0, bfr, acc[0][ni]);
      acc[1][ni] = wmma_bf16(a1, bfr, acc[1][ni]);
    }
  }

#pragma unroll
  for (int mi = 0; mi < 2; ++mi) {
#pragma unroll
    for (int ni = 0; ni < 4; ++ni) {
#pragma unroll
      for (int r = 0; r < 8; ++r) {
        int m = m0 + mi * 16 + r + 8 * hf;
        int n = n0 + ni * 16 + lm;
        out[(size_t)m * D_EMBED + n] = acc[mi][ni][r] + bout[n];
      }
    }
  }
}

// ---------- host launcher ----------
extern "C" void kernel_launch(void* const* d_in, const int* in_sizes, int n_in,
                              void* d_out, int out_size, void* d_ws, size_t ws_size,
                              hipStream_t stream) {
  const float* x     = (const float*)d_in[0];  // [2,2048,768]
  const float* w_in  = (const float*)d_in[1];  // [768,2304]
  const float* b_in  = (const float*)d_in[2];  // [2304]
  const float* w_out = (const float*)d_in[3];  // [768,768]
  const float* b_out = (const float*)d_in[4];  // [768]
  float* out = (float*)d_out;                  // [2,2048,768] fp32

  // workspace layout (bf16 elements)
  bf16_t* Xh  = (bf16_t*)d_ws;                         // 4096*768
  bf16_t* WiT = Xh  + (size_t)M_TOT * D_EMBED;         // 2304*768 (w_in^T)
  bf16_t* WoT = WiT + (size_t)D3 * D_EMBED;            // 768*768  (w_out^T)
  bf16_t* Qh  = WoT + (size_t)D_EMBED * D_EMBED;       // [24][2048][64]
  bf16_t* Kh  = Qh  + (size_t)BATCH * N_HEADS * SEQ * DH;
  bf16_t* Vt  = Kh  + (size_t)BATCH * N_HEADS * SEQ * DH;  // [24][64][2048]
  bf16_t* Ob  = Vt  + (size_t)BATCH * N_HEADS * SEQ * DH;  // [4096][768]

  const int nX  = M_TOT * D_EMBED;     // 3145728
  const int nWi = D_EMBED * D3;        // 1769472
  const int nWo = D_EMBED * D_EMBED;   // 589824

  cvt_bf16   <<<(nX  + 255) / 256, 256, 0, stream>>>(x, Xh, nX);
  cvt_bf16_tr<<<(nWi + 255) / 256, 256, 0, stream>>>(w_in,  WiT, D_EMBED, D3);
  cvt_bf16_tr<<<(nWo + 255) / 256, 256, 0, stream>>>(w_out, WoT, D_EMBED, D_EMBED);

  qkv_gemm<<<dim3(M_TOT / 32, D3 / 64), 32, 0, stream>>>(Xh, WiT, b_in, Qh, Kh, Vt);
  attn    <<<dim3(SEQ / QTILE, BATCH * N_HEADS), 128, 0, stream>>>(Qh, Kh, Vt, Ob);
  out_gemm<<<dim3(M_TOT / 32, D_EMBED / 64), 32, 0, stream>>>(Ob, WoT, b_out, out);
}
